// Tacotron_17033840296605
// MI455X (gfx1250) — compile-verified, matrix operand order checked by reading, not run on hardware
//
#include <hip/hip_runtime.h>
#include <hip/hip_bf16.h>

// ---------------------------------------------------------------------------
// Tacotron decoder for gfx1250 (MI455X): bf16 WMMA GEMMs + VALU attention.
// Weights converted per call to fragment-swizzled packed bf16; per-block the
// shared B-tile slabs are DMA'd into LDS via GLOBAL_LOAD_ASYNC_TO_LDS_B128
// (ASYNCcnt) -- 8x reduction of redundant per-wave L2 reads -- and WMMA reads
// B fragments from LDS (ds_load_b128). Activations stay f32 and are converted
// with v_cvt_pk_bf16_f32 on load; accumulation is f32 WMMA.
// ---------------------------------------------------------------------------

typedef __attribute__((ext_vector_type(16))) __bf16 v16bf;
typedef __attribute__((ext_vector_type(8)))  float  v8f;
typedef __attribute__((ext_vector_type(4)))  int    v4i;

#define B_     128
#define H_     256
#define D_     80
#define TIN_   512
#define RSTEPS 200
#define RFAC   5

#if defined(__has_builtin)
#if __has_builtin(__builtin_amdgcn_global_load_async_to_lds_b128) && \
    __has_builtin(__builtin_amdgcn_s_wait_asynccnt)
#define HAVE_ASYNC_LDS 1
#endif
#endif

// ---- async 16B global->LDS copy (falls back to load + ds_store) ------------
__device__ __forceinline__ void cp_async16(void* lds, const void* g) {
#if defined(HAVE_ASYNC_LDS)
  __builtin_amdgcn_global_load_async_to_lds_b128(
      (__attribute__((address_space(1))) v4i*)(g),
      (__attribute__((address_space(3))) v4i*)(lds), 0, 0);
#else
  *(float4*)lds = *(const float4*)g;
#endif
}
__device__ __forceinline__ void cp_commit_wait() {
#if defined(HAVE_ASYNC_LDS)
  __builtin_amdgcn_s_wait_asynccnt(0);
#endif
  __syncthreads();
}

// ---- WMMA fragment helpers -------------------------------------------------
// A fragment (16x32 bf16, MxK): lane l<16 -> row M=l, K={kb..kb+7, kb+16..kb+23},
// kb=0; lanes 16..31 -> same rows with kb=8.  Loaded from f32 row-major with
// on-the-fly packed bf16 conversion (4x b128 loads + v_cvt_pk_bf16_f32).
__device__ __forceinline__ v16bf load_a_f32(const float* __restrict__ X, int ldx,
                                            int row0, int k0) {
  int lane = threadIdx.x & 31;
  int row  = row0 + (lane & 15);
  int kb   = k0 + ((lane >> 4) << 3);
  const float* p = X + (size_t)row * ldx + kb;
  float4 f0 = *(const float4*)(p + 0);
  float4 f1 = *(const float4*)(p + 4);
  float4 f2 = *(const float4*)(p + 16);
  float4 f3 = *(const float4*)(p + 20);
  v16bf a;
  a[0]=(__bf16)f0.x;  a[1]=(__bf16)f0.y;  a[2]=(__bf16)f0.z;  a[3]=(__bf16)f0.w;
  a[4]=(__bf16)f1.x;  a[5]=(__bf16)f1.y;  a[6]=(__bf16)f1.z;  a[7]=(__bf16)f1.w;
  a[8]=(__bf16)f2.x;  a[9]=(__bf16)f2.y;  a[10]=(__bf16)f2.z; a[11]=(__bf16)f2.w;
  a[12]=(__bf16)f3.x; a[13]=(__bf16)f3.y; a[14]=(__bf16)f3.z; a[15]=(__bf16)f3.w;
  return a;
}

// B fragment (32x16 bf16, KxN) from an LDS-staged packed slab: each lane's 16
// values are contiguous (32B LDS read).
__device__ __forceinline__ v16bf load_b_lds(const char* smem, int tile) {
  int lane = threadIdx.x & 31;
  return *(const v16bf*)(smem + ((size_t)tile << 10) + ((size_t)lane << 5));
}

__device__ __forceinline__ v8f wmma_bf16(v16bf a, v16bf b, v8f c) {
  return __builtin_amdgcn_wmma_f32_16x16x32_bf16(false, a, false, b, (short)0, c,
                                                 false, false);
}

__device__ __forceinline__ float sigmoidf_(float x) {
  return 1.f / (1.f + __expf(-x));
}

// ---- weight packing: f32 [K,N] (optionally two stacked matrices) -> packed bf16
// Packed layout: tile (kt,nt) base = (nt*ktiles + kt)*512 elems; within a tile,
// lane l's 16 fragment values are contiguous.
__global__ void pack_w_kernel(const float* __restrict__ A, int Ka,
                              const float* __restrict__ Bm, int Kb,
                              int N, int Kp, __bf16* __restrict__ out,
                              long long total) {
  int ktiles = Kp >> 5;
  for (long long i = (long long)blockIdx.x * blockDim.x + threadIdx.x; i < total;
       i += (long long)gridDim.x * blockDim.x) {
    long long t = i >> 9;
    int o  = (int)(i & 511);
    int nt = (int)(t / ktiles);
    int kt = (int)(t % ktiles);
    int l  = o >> 4;
    int j  = o & 15;
    int n  = nt * 16 + (l & 15);
    int kb = (l >> 4) << 3;
    int kl = (j < 8) ? (kb + j) : (kb + 8 + j);
    int k  = kt * 32 + kl;
    float v = 0.f;
    int Kt = Ka + Kb;
    if (k < Kt)
      v = (k < Ka) ? A[(long long)k * N + n] : Bm[(long long)(k - Ka) * N + n];
    out[i] = (__bf16)v;
  }
}

// ---- generic GEMM: out = act(X[M,K] @ Wp + bias), optionally replicated ----
// block = 256 threads (8 waves); wave tile = 16 rows x 32 cols; block tile =
// 128 rows x 32 cols.  The block's B slab (2*ktiles KB, contiguous in packed
// layout) is async-DMA'd to LDS once and shared by all 8 waves.
__global__ void gemm_kernel(const float* __restrict__ X, int ldx,
                            const __bf16* __restrict__ Wp, int ktiles,
                            const float* __restrict__ bias,
                            float* __restrict__ out, int ldo, int ntn, int act,
                            int repeat, long long rstride) {
  extern __shared__ char smem[];
  int wv   = threadIdx.x >> 5;
  int lane = threadIdx.x & 31;
  int row0 = blockIdx.y * 128 + wv * 16;
  int nt0  = blockIdx.x * 2;
  bool has2 = (nt0 + 1) < ntn;

  // stage B slab: tiles [nt0*ktiles, (nt0+tiles)*ktiles) are contiguous
  int tiles = has2 ? 2 : 1;
  int slab  = tiles * ktiles * 1024;
  const char* gsrc = (const char*)Wp + (((size_t)nt0 * ktiles) << 10);
  for (int o = threadIdx.x * 16; o < slab; o += 256 * 16)
    cp_async16(smem + o, gsrc + o);
  cp_commit_wait();

  v8f acc0 = {0.f, 0.f, 0.f, 0.f, 0.f, 0.f, 0.f, 0.f};
  v8f acc1 = acc0;
  for (int kt = 0; kt < ktiles; ++kt) {
    v16bf a  = load_a_f32(X, ldx, row0, kt * 32);
    v16bf b0 = load_b_lds(smem, kt);
    acc0 = wmma_bf16(a, b0, acc0);
    if (has2) {
      v16bf b1 = load_b_lds(smem, ktiles + kt);
      acc1 = wmma_bf16(a, b1, acc1);
    }
  }
  int rbase = row0 + ((lane >> 4) << 3);
  int c0 = nt0 * 16 + (lane & 15);
#pragma unroll
  for (int i = 0; i < 8; ++i) {
    int row = rbase + i;
    float v = acc0[i] + (bias ? bias[c0] : 0.f);
    if (act) v = fmaxf(v, 0.f);
    for (int rp = 0; rp < repeat; ++rp)
      out[(size_t)rp * rstride + (size_t)row * ldo + c0] = v;
    if (has2) {
      int c1 = c0 + 16;
      float v1 = acc1[i] + (bias ? bias[c1] : 0.f);
      if (act) v1 = fmaxf(v1, 0.f);
      for (int rp = 0; rp < repeat; ++rp)
        out[(size_t)rp * rstride + (size_t)row * ldo + c1] = v1;
    }
  }
}

// ---- fused GRU cell: hnew = (1-z)*h + z*tanh(gc_x + bc + r*gc_h) -----------
// Wg packed as [(Kx+H) x 768] bf16: columns z(0..255) r(256..511) c(512..767),
// rows = [x-rows ; h-rows].  Three per-gate B slabs (each ktot KB, contiguous)
// are async-DMA'd to LDS and shared by all 8 waves; z/r accumulate over the
// concatenated K, candidate is split into x/h accumulators for r*gh_c.
// grid.x = 16 (column tiles of H), block = 256 threads (rows 0..127).
__global__ void gru_kernel(const float* __restrict__ X, int ldx, int Kx,
                           const float* __restrict__ Hin,
                           const __bf16* __restrict__ Wg,
                           const float* __restrict__ b3h,
                           float* __restrict__ Hout,
                           const float* __restrict__ Resin,
                           float* __restrict__ Resout,
                           float* __restrict__ Mirror, int ldm) {
  extern __shared__ char smem[];
  int wv   = threadIdx.x >> 5;
  int lane = threadIdx.x & 31;
  int row0 = wv * 16;
  int c    = blockIdx.x;
  int ktot = (Kx + H_) >> 5;
  int kxt  = Kx >> 5;

  // stage the three gate slabs: global tiles (gt*16 + c, 0..ktot-1) contiguous
  int slab = ktot * 1024;
  for (int gt = 0; gt < 3; ++gt) {
    const char* src = (const char*)Wg + (((size_t)(gt * 16 + c) * ktot) << 10);
    char* dst = smem + (size_t)gt * slab;
    for (int o = threadIdx.x * 16; o < slab; o += 256 * 16)
      cp_async16(dst + o, src + o);
  }
  cp_commit_wait();

  v8f az = {0.f, 0.f, 0.f, 0.f, 0.f, 0.f, 0.f, 0.f};
  v8f ar = az, acx = az, ach = az;
  for (int kt = 0; kt < ktot; ++kt) {
    int k0 = kt * 32;
    const float* src;
    int ld, kk;
    if (k0 < Kx) { src = X;   ld = ldx; kk = k0; }
    else         { src = Hin; ld = H_;  kk = k0 - Kx; }
    v16bf a  = load_a_f32(src, ld, row0, kk);
    v16bf bz = load_b_lds(smem, kt);
    v16bf br = load_b_lds(smem, ktot + kt);
    v16bf bc = load_b_lds(smem, 2 * ktot + kt);
    az = wmma_bf16(a, bz, az);
    ar = wmma_bf16(a, br, ar);
    if (kt < kxt) acx = wmma_bf16(a, bc, acx);
    else          ach = wmma_bf16(a, bc, ach);
  }
  int rbase = row0 + ((lane >> 4) << 3);
  int col = c * 16 + (lane & 15);
  float bzv = b3h[col], brv = b3h[H_ + col], bcv = b3h[2 * H_ + col];
#pragma unroll
  for (int i = 0; i < 8; ++i) {
    int row = rbase + i;
    float z = sigmoidf_(az[i] + bzv);
    float r = sigmoidf_(ar[i] + brv);
    float cc = tanhf(acx[i] + bcv + r * ach[i]);
    float hold = Hin[(size_t)row * H_ + col];
    float hn = (1.f - z) * hold + z * cc;
    Hout[(size_t)row * H_ + col] = hn;
    if (Resout)
      Resout[(size_t)row * H_ + col] = Resin[(size_t)row * H_ + col] + hn;
    if (Mirror)
      Mirror[(size_t)row * ldm + col] = hn;
  }
}

// ---- additive attention: one block per batch element -----------------------
// e[t] = sum_h v[h]*tanh(keys[t,b,h] + qk[b,h]); softmax over t; context and
// alpha written out.  keys/memory (134MB) are L2-resident (192MB L2).
__global__ void attention_kernel(const float* __restrict__ keys,
                                 const float* __restrict__ qk,
                                 const float* __restrict__ vatt,
                                 const float* __restrict__ memory,
                                 float* __restrict__ xcat,
                                 float* __restrict__ alpha_out) {
  __shared__ float e[TIN_];
  __shared__ float red[16];
  int b    = blockIdx.x;
  int tid  = threadIdx.x;
  int lane = tid & 31;
  int wv   = tid >> 5;
  const float* qb = qk + (size_t)b * H_ + lane * 8;
  float4 q0 = *(const float4*)(qb);
  float4 q1 = *(const float4*)(qb + 4);
  float4 v0 = *(const float4*)(vatt + lane * 8);
  float4 v1 = *(const float4*)(vatt + lane * 8 + 4);
  for (int t = wv; t < TIN_; t += 8) {
    const float* kr = keys + ((size_t)t * B_ + b) * H_ + lane * 8;
    float4 k0 = *(const float4*)(kr);
    float4 k1 = *(const float4*)(kr + 4);
    float s = v0.x * tanhf(k0.x + q0.x) + v0.y * tanhf(k0.y + q0.y) +
              v0.z * tanhf(k0.z + q0.z) + v0.w * tanhf(k0.w + q0.w) +
              v1.x * tanhf(k1.x + q1.x) + v1.y * tanhf(k1.y + q1.y) +
              v1.z * tanhf(k1.z + q1.z) + v1.w * tanhf(k1.w + q1.w);
    for (int off = 16; off; off >>= 1) s += __shfl_xor(s, off, 32);
    if (lane == 0) e[t] = s;
  }
  __syncthreads();
  float m = fmaxf(e[tid], e[tid + 256]);
  for (int off = 16; off; off >>= 1) m = fmaxf(m, __shfl_xor(m, off, 32));
  if (lane == 0) red[wv] = m;
  __syncthreads();
  float mx = red[0];
#pragma unroll
  for (int w = 1; w < 8; ++w) mx = fmaxf(mx, red[w]);
  float ex0 = __expf(e[tid] - mx);
  float ex1 = __expf(e[tid + 256] - mx);
  float ss = ex0 + ex1;
  for (int off = 16; off; off >>= 1) ss += __shfl_xor(ss, off, 32);
  if (lane == 0) red[8 + wv] = ss;
  __syncthreads();
  float ssum = red[8];
#pragma unroll
  for (int w = 1; w < 8; ++w) ssum += red[8 + w];
  float a0 = ex0 / ssum, a1 = ex1 / ssum;
  e[tid] = a0;
  e[tid + 256] = a1;
  alpha_out[(size_t)b * TIN_ + tid] = a0;
  alpha_out[(size_t)b * TIN_ + 256 + tid] = a1;
  __syncthreads();
  float ctx = 0.f;
  for (int t = 0; t < TIN_; ++t)
    ctx += e[t] * memory[((size_t)t * B_ + b) * H_ + tid];
  xcat[(size_t)b * 512 + 256 + tid] = ctx;  // second half of GRU0 input
}

// ---- teacher-forcing frame, zero-padded from D=80 to K=96 ------------------
__global__ void prep_frame_kernel(const float* __restrict__ outputs,
                                  float* __restrict__ xf, int step) {
  int i = blockIdx.x * blockDim.x + threadIdx.x;
  if (i >= B_ * 96) return;
  int b = i / 96, k = i % 96;
  float v = 0.f;
  if (k < D_ && step > 0)
    v = outputs[((size_t)(step * RFAC - 1) * B_ + b) * D_ + k];
  xf[i] = v;
}

__global__ void zero_kernel(float* __restrict__ p, int n) {
  int i = blockIdx.x * blockDim.x + threadIdx.x;
  if (i < n) p[i] = 0.f;
}

// ---------------------------------------------------------------------------
extern "C" void kernel_launch(void* const* d_in, const int* in_sizes, int n_in,
                              void* d_out, int out_size, void* d_ws, size_t ws_size,
                              hipStream_t stream) {
  (void)in_sizes; (void)n_in; (void)out_size;
  const float* outputs = (const float*)d_in[0];
  const float* memory  = (const float*)d_in[1];
  const float* W_pre0  = (const float*)d_in[2];
  const float* b_pre0  = (const float*)d_in[3];
  const float* W_pre1  = (const float*)d_in[4];
  const float* b_pre1  = (const float*)d_in[5];
  const float* Wx_att  = (const float*)d_in[6];
  const float* Wh_att  = (const float*)d_in[7];
  const float* b_att   = (const float*)d_in[8];
  const float* Wq      = (const float*)d_in[9];
  const float* Wm      = (const float*)d_in[10];
  const float* v_att   = (const float*)d_in[11];
  const float* Wx_d0   = (const float*)d_in[12];
  const float* Wh_d0   = (const float*)d_in[13];
  const float* b_d0    = (const float*)d_in[14];
  const float* Wx_d1   = (const float*)d_in[15];
  const float* Wh_d1   = (const float*)d_in[16];
  const float* b_d1    = (const float*)d_in[17];
  const float* Wo      = (const float*)d_in[18];
  const float* bo      = (const float*)d_in[19];

  float* out       = (float*)d_out;
  float* alpha_out = out + (size_t)1000 * B_ * D_;  // [R,B,T_in]

  char* ws = (char*)d_ws;
  size_t off = 0;
  auto take = [&](size_t bytes) -> char* {
    char* p = ws + off;
    off += (bytes + 255) & ~(size_t)255;
    return p;
  };
  float*  keys  = (float*)take((size_t)TIN_ * B_ * H_ * 4);   // 67.1 MB
  __bf16* pre0p = (__bf16*)take((size_t)96 * 256 * 2);
  __bf16* pre1p = (__bf16*)take((size_t)256 * 128 * 2);
  __bf16* attgp = (__bf16*)take((size_t)384 * 768 * 2);
  __bf16* d0gp  = (__bf16*)take((size_t)768 * 768 * 2);
  __bf16* d1gp  = (__bf16*)take((size_t)512 * 768 * 2);
  __bf16* wqp   = (__bf16*)take((size_t)256 * 256 * 2);
  __bf16* wmp   = (__bf16*)take((size_t)256 * 256 * 2);
  __bf16* wop   = (__bf16*)take((size_t)256 * 80 * 2);
  float* xf   = (float*)take((size_t)B_ * 96 * 4);
  float* hid  = (float*)take((size_t)B_ * 256 * 4);
  float* pre  = (float*)take((size_t)B_ * 128 * 4);
  float* hatt[2] = {(float*)take((size_t)B_ * H_ * 4), (float*)take((size_t)B_ * H_ * 4)};
  float* hd0[2]  = {(float*)take((size_t)B_ * H_ * 4), (float*)take((size_t)B_ * H_ * 4)};
  float* hd1[2]  = {(float*)take((size_t)B_ * H_ * 4), (float*)take((size_t)B_ * H_ * 4)};
  float* qk   = (float*)take((size_t)B_ * H_ * 4);
  float* xcat = (float*)take((size_t)B_ * 512 * 4);  // [query | context]
  float* res0 = (float*)take((size_t)B_ * H_ * 4);
  float* res1 = (float*)take((size_t)B_ * H_ * 4);
  if (off > ws_size) return;  // workspace too small: nothing safe to do

  auto pack = [&](const float* A, int Ka, const float* Bm, int Kb, int N, int Kp,
                  __bf16* dst) {
    long long total = (long long)Kp * N;
    int blocks = (int)((total + 255) / 256);
    pack_w_kernel<<<blocks, 256, 0, stream>>>(A, Ka, Bm, Kb, N, Kp, dst, total);
  };
  pack(W_pre0, 80, nullptr, 0, 256, 96, pre0p);
  pack(W_pre1, 256, nullptr, 0, 128, 256, pre1p);
  pack(Wx_att, 128, Wh_att, 256, 768, 384, attgp);
  pack(Wx_d0, 512, Wh_d0, 256, 768, 768, d0gp);
  pack(Wx_d1, 256, Wh_d1, 256, 768, 512, d1gp);
  pack(Wq, 256, nullptr, 0, 256, 256, wqp);
  pack(Wm, 256, nullptr, 0, 256, 256, wmp);
  pack(Wo, 256, nullptr, 0, 80, 256, wop);

  // attention keys once: [T_in*B, H] @ Wm  (8.6 GFLOP, WMMA)
  gemm_kernel<<<dim3(8, (TIN_ * B_) / 128), 256, 2 * 8 * 1024, stream>>>(
      memory, H_, wmp, 8, nullptr, keys, H_, 16, 0, 1, 0);

  zero_kernel<<<(B_ * H_ + 255) / 256, 256, 0, stream>>>(hatt[0], B_ * H_);
  zero_kernel<<<(B_ * H_ + 255) / 256, 256, 0, stream>>>(hd0[0], B_ * H_);
  zero_kernel<<<(B_ * H_ + 255) / 256, 256, 0, stream>>>(hd1[0], B_ * H_);

  for (int s = 0; s < RSTEPS; ++s) {
    float* ha_in  = hatt[s & 1];
    float* ha_out = hatt[(s + 1) & 1];
    float* h0_in  = hd0[s & 1];
    float* h0_out = hd0[(s + 1) & 1];
    float* h1_in  = hd1[s & 1];
    float* h1_out = hd1[(s + 1) & 1];

    prep_frame_kernel<<<(B_ * 96 + 255) / 256, 256, 0, stream>>>(outputs, xf, s);
    // prenet: relu(frame@W0+b0), relu(.@W1+b1)
    gemm_kernel<<<dim3(8, 1), 256, 2 * 3 * 1024, stream>>>(
        xf, 96, pre0p, 3, b_pre0, hid, 256, 16, 1, 1, 0);
    gemm_kernel<<<dim3(4, 1), 256, 2 * 8 * 1024, stream>>>(
        hid, 256, pre1p, 8, b_pre1, pre, 128, 8, 1, 1, 0);
    // attention GRU -> query (also mirrored into xcat[:, 0:256])
    gru_kernel<<<16, 256, 3 * 12 * 1024, stream>>>(
        pre, 128, 128, ha_in, attgp, b_att, ha_out, nullptr, nullptr, xcat, 512);
    // q @ Wq
    gemm_kernel<<<dim3(8, 1), 256, 2 * 8 * 1024, stream>>>(
        ha_out, H_, wqp, 8, nullptr, qk, H_, 16, 0, 1, 0);
    // additive attention: alpha -> d_out tail, context -> xcat[:, 256:512]
    attention_kernel<<<B_, 256, 0, stream>>>(
        keys, qk, v_att, memory, xcat, alpha_out + (size_t)s * B_ * TIN_);
    // decoder GRU0 on [query|context]; res0 = query + out0
    gru_kernel<<<16, 256, 3 * 24 * 1024, stream>>>(
        xcat, 512, 512, h0_in, d0gp, b_d0, h0_out, ha_out, res0, nullptr, 0);
    // decoder GRU1 on res0; res1 = res0 + out1
    gru_kernel<<<16, 256, 3 * 16 * 1024, stream>>>(
        res0, 256, 256, h1_in, d1gp, b_d1, h1_out, res0, res1, nullptr, 0);
    // dense output, replicated into r=5 consecutive frames
    gemm_kernel<<<dim3(3, 1), 256, 2 * 8 * 1024, stream>>>(
        res1, 256, wop, 8, bo, out + (size_t)s * RFAC * B_ * D_, D_, 5, 0, 5,
        (long long)B_ * D_);
  }
}
